// MHAttention_60215441490286
// MI455X (gfx1250) — matrix operand order repaired and checked by reading, not compiled, not run on hardware
//
#include <hip/hip_runtime.h>

// ---------------------------------------------------------------------------
// MHA (no-softmax variant): out = scale * Q @ (K^T @ V), per batch*head.
// All GEMMs run on v_wmma_f32_16x16x32_bf16 (bf16 in, fp32 accumulate).
// ---------------------------------------------------------------------------

typedef __bf16 v8bf  __attribute__((ext_vector_type(8)));
typedef __bf16 v16bf __attribute__((ext_vector_type(16)));
typedef float  v8f   __attribute__((ext_vector_type(8)));

#define E_DIM   1024
#define M_TOT   4096      // b*s = 2*2048 tokens
#define SEQ_B   2048      // per-batch-head rows in the reshaped view
#define HD      64        // head dim
#define NBATCH  32        // b*h

__device__ __forceinline__ unsigned short f2bf(float f) {
  union { float f; unsigned u; } v; v.f = f;
  unsigned u = v.u;
  u += 0x7FFFu + ((u >> 16) & 1u);   // round to nearest even
  return (unsigned short)(u >> 16);
}

__device__ __forceinline__ unsigned pack2bf(float lo, float hi) {
  return (unsigned)f2bf(lo) | ((unsigned)f2bf(hi) << 16);
}

// A fragment (16x32 bf16, M x K): lane row = lane&15, K halves split by lane>>4.
__device__ __forceinline__ v16bf load_frag_a(const unsigned short* row_ptr, int k, int lane) {
  int ko = k + ((lane >> 4) << 3);
  v8bf lo = *(const v8bf*)(row_ptr + ko);
  v8bf hi = *(const v8bf*)(row_ptr + ko + 16);
  v16bf r;
#pragma unroll
  for (int i = 0; i < 8; ++i) { r[i] = lo[i]; r[i + 8] = hi[i]; }
  return r;
}

// B fragment (32x16 bf16, K x N), sourced from an "N x K row-major" buffer:
// lane column n = lane&15; K halves split by lane>>4 (8 at k+(lane>>4)*8, 8 at +16).
__device__ __forceinline__ v16bf load_frag_b(const unsigned short* row_ptr, int k, int lane) {
  int ko = k + ((lane >> 4) << 3);
  v8bf lo = *(const v8bf*)(row_ptr + ko);
  v8bf hi = *(const v8bf*)(row_ptr + ko + 16);
  v16bf r;
#pragma unroll
  for (int i = 0; i < 8; ++i) { r[i] = lo[i]; r[i + 8] = hi[i]; }
  return r;
}

__device__ __forceinline__ v8f wmma_bf16(v16bf a, v16bf b, v8f c) {
  return __builtin_amdgcn_wmma_f32_16x16x32_bf16(
      /*neg_a=*/false, a, /*neg_b=*/false, b,
      /*c_mod=*/(short)0, c, /*reuse_a=*/false, /*reuse_b=*/false);
}

// ---------------------------------------------------------------------------
// Kernel 1: fp32 -> bf16, 8 elements per thread (b128 in/out).
// ---------------------------------------------------------------------------
__global__ __launch_bounds__(256) void cvt_f32_bf16(const float* __restrict__ src,
                                                    unsigned short* __restrict__ dst,
                                                    int n8) {
  int i = blockIdx.x * blockDim.x + threadIdx.x;
  if (i >= n8) return;
  float4 v0 = ((const float4*)src)[2 * i];
  float4 v1 = ((const float4*)src)[2 * i + 1];
  uint4 o;
  o.x = pack2bf(v0.x, v0.y);
  o.y = pack2bf(v0.z, v0.w);
  o.z = pack2bf(v1.x, v1.y);
  o.w = pack2bf(v1.z, v1.w);
  ((uint4*)dst)[i] = o;
}

// ---------------------------------------------------------------------------
// Kernel 2: QKV projection. y = x @ W^T + b   (TN GEMM: M=4096, N=1024, K=1024)
// Workgroup: 512 threads = 16 waves; tile 16 rows x FULL N=1024
// (each wave 16x64, acc[4] -> low register pressure so the double-buffered
//  next-K fragments stay live and overlap with the WMMAs).
// Full-width tiles make the per-WG transposed K/V chunk contiguous:
//   Kt[batch][d][j0 .. j0+256)  for all d in 0..63  (512B runs)
// staged through 32KB LDS and stored fully coalesced.
//   z=0 -> Q stored bf16 flat [4096,1024] (== [32,2048,64] reshape)
//   z=1 -> Kt[batch][d][j], z=2 -> Vt[batch][d][j]   ([64,2048] per batch-head)
// ---------------------------------------------------------------------------
__global__ __launch_bounds__(512) void qkv_gemm(const unsigned short* __restrict__ Xbf,
                                                const unsigned short* __restrict__ Wbf,
                                                const float* __restrict__ bq,
                                                const float* __restrict__ bk,
                                                const float* __restrict__ bv,
                                                unsigned short* __restrict__ Qbf,
                                                unsigned short* __restrict__ Ktb,
                                                unsigned short* __restrict__ Vtb) {
  __shared__ unsigned short lds_t[64][256];   // 32KB transpose staging (z=1,2)

  const int z    = blockIdx.y;
  const unsigned short* W = Wbf + (size_t)z * E_DIM * E_DIM;
  const float* bias = (z == 0) ? bq : (z == 1) ? bk : bv;

  const int lane = threadIdx.x & 31;
  const int w    = threadIdx.x >> 5;    // 16 waves
  const int lrow = lane & 15;
  const int m0   = blockIdx.x * 16;     // 16 output rows per WG
  const int nb   = w * 64;              // 64 output cols per wave

  v8f acc[4];
#pragma unroll
  for (int ni = 0; ni < 4; ++ni)
#pragma unroll
    for (int r = 0; r < 8; ++r) acc[ni][r] = 0.0f;

  const unsigned short* arow = Xbf + (size_t)(m0 + lrow) * E_DIM;
  const unsigned short* brow[4];
#pragma unroll
  for (int ni = 0; ni < 4; ++ni)
    brow[ni] = W + (size_t)(nb + ni * 16 + lrow) * E_DIM;

  // ---- software-pipelined K loop (double-buffered fragments) ----
  v16bf a_cur = load_frag_a(arow, 0, lane);
  v16bf b_cur[4];
#pragma unroll
  for (int ni = 0; ni < 4; ++ni) b_cur[ni] = load_frag_b(brow[ni], 0, lane);

  for (int k = 0; k < E_DIM - 32; k += 32) {
    if (k + 256 < E_DIM) {   // stay ~8 steps ahead; per-lane row prefetch
      __builtin_prefetch(arow + k + 256, 0, 1);
      __builtin_prefetch(brow[0] + k + 256, 0, 1);
      __builtin_prefetch(brow[1] + k + 256, 0, 1);
      __builtin_prefetch(brow[2] + k + 256, 0, 1);
      __builtin_prefetch(brow[3] + k + 256, 0, 1);
    }
    // issue next-step loads (independent regs), then drain current WMMAs
    v16bf a_nxt = load_frag_a(arow, k + 32, lane);
    v16bf b_nxt[4];
#pragma unroll
    for (int ni = 0; ni < 4; ++ni) b_nxt[ni] = load_frag_b(brow[ni], k + 32, lane);
#pragma unroll
    for (int ni = 0; ni < 4; ++ni) acc[ni] = wmma_bf16(a_cur, b_cur[ni], acc[ni]);
    a_cur = a_nxt;
#pragma unroll
    for (int ni = 0; ni < 4; ++ni) b_cur[ni] = b_nxt[ni];
  }
#pragma unroll
  for (int ni = 0; ni < 4; ++ni) acc[ni] = wmma_bf16(a_cur, b_cur[ni], acc[ni]);

  // ---- epilogue ----
  if (z == 0) {
    // Q: direct bf16 stores; lanes 0-15 cover 32B-contiguous runs.
#pragma unroll
    for (int ni = 0; ni < 4; ++ni) {
#pragma unroll
      for (int r = 0; r < 8; ++r) {
        int mg = m0 + ((lane >> 4) << 3) + r;
        int ng = nb + ni * 16 + lrow;
        Qbf[(size_t)mg * E_DIM + ng] = f2bf(acc[ni][r] + bias[ng]);
      }
    }
  } else {
    // K/V: transpose through LDS, then coalesced 512B-row stores.
#pragma unroll
    for (int ni = 0; ni < 4; ++ni) {
#pragma unroll
      for (int r = 0; r < 8; ++r) {
        int mg = m0 + ((lane >> 4) << 3) + r;
        int ng = nb + ni * 16 + lrow;
        // global: batch = mg>>7, j = ((mg&127)<<4)|(ng>>6), d = ng&63
        lds_t[ng & 63][((mg & 15) << 4) | (ng >> 6)] = f2bf(acc[ni][r] + bias[ng]);
      }
    }
    __syncthreads();

    const int batch = m0 >> 7;
    const int j0    = (m0 & 127) << 4;
    unsigned short* dst = ((z == 1) ? Ktb : Vtb) + ((size_t)batch << 17);
    const int d = threadIdx.x >> 3;          // 64 rows
    const int c = (threadIdx.x & 7) * 32;    // 8 x 64B chunks per row
    const uint4* s = (const uint4*)&lds_t[d][c];
    uint4* g = (uint4*)(dst + ((size_t)d << 11) + j0 + c);
#pragma unroll
    for (int i = 0; i < 4; ++i) g[i] = s[i];
  }
}

// ---------------------------------------------------------------------------
// Kernel 3: S = scale * K^T V per batch-head.  TN GEMM: M=64, N=64, K=2048.
// One workgroup per batch; 8 waves, each 16x32 (2 accs), double-buffered.
// Output stored TRANSPOSED (St[d][kk]) + scaled + bf16 so stage 4 is TN again.
// ---------------------------------------------------------------------------
__global__ __launch_bounds__(256) void kv_reduce(const unsigned short* __restrict__ Ktb,
                                                 const unsigned short* __restrict__ Vtb,
                                                 unsigned short* __restrict__ St) {
  const int batch = blockIdx.x;
  const unsigned short* Kt = Ktb + ((size_t)batch << 17);
  const unsigned short* Vt = Vtb + ((size_t)batch << 17);

  const int lane = threadIdx.x & 31;
  const int w    = threadIdx.x >> 5;
  const int lrow = lane & 15;
  const int m0 = (w >> 1) * 16;   // kk (K-feature dim)
  const int n0 = (w & 1) * 32;    // d  (V-feature dim)

  v8f acc[2];
#pragma unroll
  for (int ni = 0; ni < 2; ++ni)
#pragma unroll
    for (int r = 0; r < 8; ++r) acc[ni][r] = 0.0f;

  const unsigned short* arow  = Kt + (size_t)(m0 + lrow) * SEQ_B;
  const unsigned short* brow0 = Vt + (size_t)(n0 + lrow) * SEQ_B;
  const unsigned short* brow1 = brow0 + (size_t)16 * SEQ_B;

  v16bf a_cur  = load_frag_a(arow, 0, lane);
  v16bf b0_cur = load_frag_b(brow0, 0, lane);
  v16bf b1_cur = load_frag_b(brow1, 0, lane);

  for (int k = 0; k < SEQ_B - 32; k += 32) {
    if (k + 256 < SEQ_B) {
      __builtin_prefetch(arow + k + 256, 0, 1);
      __builtin_prefetch(brow0 + k + 256, 0, 1);
      __builtin_prefetch(brow1 + k + 256, 0, 1);
    }
    v16bf a_nxt  = load_frag_a(arow, k + 32, lane);
    v16bf b0_nxt = load_frag_b(brow0, k + 32, lane);
    v16bf b1_nxt = load_frag_b(brow1, k + 32, lane);
    acc[0] = wmma_bf16(a_cur, b0_cur, acc[0]);
    acc[1] = wmma_bf16(a_cur, b1_cur, acc[1]);
    a_cur = a_nxt; b0_cur = b0_nxt; b1_cur = b1_nxt;
  }
  acc[0] = wmma_bf16(a_cur, b0_cur, acc[0]);
  acc[1] = wmma_bf16(a_cur, b1_cur, acc[1]);

  const float scale = 0.125f;  // 1/sqrt(64)
#pragma unroll
  for (int ni = 0; ni < 2; ++ni) {
#pragma unroll
    for (int r = 0; r < 8; ++r) {
      int mg = m0 + ((lane >> 4) << 3) + r;   // kk
      int ng = n0 + ni * 16 + lrow;           // d
      St[((size_t)batch << 12) + (size_t)ng * HD + mg] = f2bf(acc[ni][r] * scale);
    }
  }
}

// ---------------------------------------------------------------------------
// Kernel 4: out = Q @ S per batch-head. TN GEMM: M=2048, N=64, K=64.
// 8 waves x 32 rows = 256 rows per WG; grid (8, 32). fp32 output.
// ---------------------------------------------------------------------------
__global__ __launch_bounds__(256) void out_gemm(const unsigned short* __restrict__ Qbf,
                                                const unsigned short* __restrict__ St,
                                                float* __restrict__ out) {
  const int batch = blockIdx.y;
  const unsigned short* Q = Qbf + ((size_t)batch << 17);
  const unsigned short* S = St + ((size_t)batch << 12);
  float* O = out + ((size_t)batch << 17);

  const int lane = threadIdx.x & 31;
  const int w    = threadIdx.x >> 5;
  const int lrow = lane & 15;
  const int m_base = blockIdx.x * 256 + w * 32;

  v8f acc[2][4];
#pragma unroll
  for (int mi = 0; mi < 2; ++mi)
#pragma unroll
    for (int ni = 0; ni < 4; ++ni)
#pragma unroll
      for (int r = 0; r < 8; ++r) acc[mi][ni][r] = 0.0f;

  const unsigned short* arow0 = Q + (size_t)(m_base + lrow) * HD;
  const unsigned short* arow1 = arow0 + (size_t)16 * HD;
  const unsigned short* brow[4];
#pragma unroll
  for (int ni = 0; ni < 4; ++ni)
    brow[ni] = S + (size_t)(ni * 16 + lrow) * HD;

#pragma unroll
  for (int k = 0; k < HD; k += 32) {
    v16bf a0 = load_frag_a(arow0, k, lane);
    v16bf a1 = load_frag_a(arow1, k, lane);
    v16bf b[4];
#pragma unroll
    for (int ni = 0; ni < 4; ++ni) b[ni] = load_frag_b(brow[ni], k, lane);
#pragma unroll
    for (int ni = 0; ni < 4; ++ni) {
      acc[0][ni] = wmma_bf16(a0, b[ni], acc[0][ni]);
      acc[1][ni] = wmma_bf16(a1, b[ni], acc[1][ni]);
    }
  }

#pragma unroll
  for (int mi = 0; mi < 2; ++mi) {
#pragma unroll
    for (int ni = 0; ni < 4; ++ni) {
#pragma unroll
      for (int r = 0; r < 8; ++r) {
        int mg = m_base + mi * 16 + ((lane >> 4) << 3) + r;  // j
        int ng = ni * 16 + lrow;                             // d
        O[(size_t)mg * HD + ng] = acc[mi][ni][r];
      }
    }
  }
}

// ---------------------------------------------------------------------------
// Launch
// ---------------------------------------------------------------------------
extern "C" void kernel_launch(void* const* d_in, const int* in_sizes, int n_in,
                              void* d_out, int out_size, void* d_ws, size_t ws_size,
                              hipStream_t stream) {
  const float* x  = (const float*)d_in[0];
  const float* Wq = (const float*)d_in[1];
  const float* bq = (const float*)d_in[2];
  const float* Wk = (const float*)d_in[3];
  const float* bk = (const float*)d_in[4];
  const float* Wv = (const float*)d_in[5];
  const float* bv = (const float*)d_in[6];
  float* out = (float*)d_out;

  const size_t nX = (size_t)M_TOT * E_DIM;       // 4,194,304
  const size_t nW = (size_t)E_DIM * E_DIM;       // 1,048,576

  char* ws = (char*)d_ws;
  unsigned short* Xbf = (unsigned short*)(ws);                       // 8 MB
  unsigned short* Wbf = (unsigned short*)(ws + nX * 2);              // 6 MB (3x2MB)
  unsigned short* Qbf = (unsigned short*)(ws + nX * 2 + 3 * nW * 2); // 8 MB
  unsigned short* Ktb = Qbf + nX;                                    // 8 MB
  unsigned short* Vtb = Ktb + nX;                                    // 8 MB
  unsigned short* St  = Vtb + nX;                                    // 256 KB

  // 1) fp32 -> bf16 conversions (8 elems/thread)
  cvt_f32_bf16<<<dim3((unsigned)(nX / 8 / 256)), dim3(256), 0, stream>>>(x, Xbf, (int)(nX / 8));
  cvt_f32_bf16<<<dim3((unsigned)(nW / 8 / 256)), dim3(256), 0, stream>>>(Wq, Wbf,          (int)(nW / 8));
  cvt_f32_bf16<<<dim3((unsigned)(nW / 8 / 256)), dim3(256), 0, stream>>>(Wk, Wbf + nW,     (int)(nW / 8));
  cvt_f32_bf16<<<dim3((unsigned)(nW / 8 / 256)), dim3(256), 0, stream>>>(Wv, Wbf + 2 * nW, (int)(nW / 8));

  // 2) QKV projections: 16-row x full-width tiles; 512 threads; grid (M/16, z)
  qkv_gemm<<<dim3(M_TOT / 16, 3), dim3(512), 0, stream>>>(
      Xbf, Wbf, bq, bk, bv, Qbf, Ktb, Vtb);

  // 3) S = scale * K^T V per batch-head
  kv_reduce<<<dim3(NBATCH), dim3(256), 0, stream>>>(Ktb, Vtb, St);

  // 4) out = Q @ S
  out_gemm<<<dim3(SEQ_B / 256, NBATCH), dim3(256), 0, stream>>>(Qbf, St, out);
}
